// BackboneGCN_22909355557430
// MI455X (gfx1250) — matrix operand order repaired and checked by reading, not compiled, not run on hardware
//
#include <hip/hip_runtime.h>
#include <hip/hip_bf16.h>
#include <cstdint>

// ---------------------------------------------------------------------------
// Problem constants (from reference)
// ---------------------------------------------------------------------------
#define N_NODES 100000
#define N_EDGES 1600000
#define NFEAT   512
#define NHID    128
#define NCLASS  64

typedef __attribute__((ext_vector_type(16))) __bf16 v16bf;
typedef __attribute__((ext_vector_type(8)))  __bf16 v8bf;
typedef __attribute__((ext_vector_type(8)))  float  v8f;
typedef __attribute__((ext_vector_type(4)))  int    i32x4;

// address-space-qualified int4 for the async global->LDS builtin
typedef __attribute__((address_space(1))) i32x4 as1_i32x4;   // global
typedef __attribute__((address_space(3))) i32x4 as3_i32x4;   // LDS

// f32 -> bf16: native hardware convert (RNE fptrunc)
__device__ __forceinline__ __bf16 f2bf(float f) { return (__bf16)f; }

__device__ __forceinline__ uint32_t mix_hash(uint32_t x) {
    x ^= 42u * 0x9E3779B9u;       // fold in "key 42"
    x ^= x >> 16; x *= 0x7FEB352Du;
    x ^= x >> 15; x *= 0x846CA68Bu;
    x ^= x >> 16;
    return x;
}

// ---------------------------------------------------------------------------
// CDNA5 async global->LDS copy (ASYNCcnt-tracked); builtin existence guarded.
// ---------------------------------------------------------------------------
#if defined(__gfx1250__) && __has_builtin(__builtin_amdgcn_global_load_async_to_lds_b128)
#define HAVE_ASYNC_LDS 1
#else
#define HAVE_ASYNC_LDS 0
#endif

#if HAVE_ASYNC_LDS
__device__ __forceinline__ void async_copy_b128(const void* gsrc, void* ldst) {
    __builtin_amdgcn_global_load_async_to_lds_b128(
        (as1_i32x4*)gsrc,      // global source (AS1 int4*)
        (as3_i32x4*)ldst,      // LDS destination (AS3 int4*)
        /*offset=*/0, /*cpol=*/0);
}
__device__ __forceinline__ void wait_async_zero() {
#if __has_builtin(__builtin_amdgcn_s_wait_asynccnt)
    __builtin_amdgcn_s_wait_asynccnt(0);
#else
    asm volatile("s_wait_asynccnt 0x0" ::: "memory");
#endif
}
#endif

// ---------------------------------------------------------------------------
// Pre-pack W[K,N] (f32) into bf16 arranged exactly as the WMMA B fragment:
//   Wpk[kstep][ntile][lane][i]  (i = 0..15 contiguous, 32B per lane)
// where lane = kh*16 + (n&15), element i corresponds to K = kstep*32+kh*16+i.
// ---------------------------------------------------------------------------
template <int K, int N>
__global__ void gcn_pack_w(const float* __restrict__ W, __bf16* __restrict__ Wpk) {
    const int gid = blockIdx.x * blockDim.x + threadIdx.x;
    if (gid >= K * N) return;
    const int k = gid / N, n = gid % N;
    const int kstep = k >> 5;
    const int ntile = n >> 4;
    const int kh    = (k >> 4) & 1;
    const int i     = k & 15;
    const int lane  = kh * 16 + (n & 15);
    Wpk[(size_t)((kstep * (N / 16) + ntile) * 32 + lane) * 16 + i] = f2bf(W[gid]);
}

// ---------------------------------------------------------------------------
// Dense GEMM via V_WMMA_F32_16X16X32_BF16.
//   out[M,N] = A[M,K] @ W[K,N] + bias[N]
// A (f32) staged to LDS: async global->LDS raw copy (16xK tile is contiguous)
// then in-LDS native bf16 conversion; B read pre-packed bf16 (32B/lane).
// ---------------------------------------------------------------------------
template <int K, int N, int NWAVES>
__global__ __launch_bounds__(NWAVES * 32)
void gcn_gemm_wmma(const float* __restrict__ A,
                   const __bf16* __restrict__ Wpk,
                   const float* __restrict__ bias,
                   float* __restrict__ out) {
    constexpr int LDA = K + 8;                  // bf16 elems; row pitch 16B-aligned
    constexpr int NT  = NWAVES * 32;
    constexpr int CV  = (16 * K) / NT;          // f32 elems converted per thread
    __shared__ __align__(128) __bf16 aTile[16 * LDA];
#if HAVE_ASYNC_LDS
    __shared__ __align__(128) float  aStage[16 * K];
#endif

    const int m0 = blockIdx.x * 16;
    const float* __restrict__ tileSrc = A + (size_t)m0 * K;   // contiguous 16*K f32

#if HAVE_ASYNC_LDS
    {   // ---- async raw copy of the whole tile into LDS (ASYNCcnt) ----
        constexpr int NB128 = (16 * K * 4) / NT / 16;         // b128 ops per thread
        #pragma unroll
        for (int j = 0; j < NB128; ++j) {
            const size_t byteOff = ((size_t)threadIdx.x * NB128 + j) * 16;
            async_copy_b128((const char*)tileSrc + byteOff, (char*)aStage + byteOff);
        }
        wait_async_zero();
        __syncthreads();
        // ---- convert f32 -> bf16 into the padded compute tile ----
        const int idx = threadIdx.x * CV;
        const int r = idx / K, c0 = idx % K;
        #pragma unroll
        for (int j = 0; j < CV; ++j)
            aTile[r * LDA + c0 + j] = f2bf(aStage[idx + j]);
    }
#else
    {   // ---- fallback: direct load + convert ----
        const int idx = threadIdx.x * CV;
        const int r = idx / K, c0 = idx % K;
        const float* src = tileSrc + idx;
        #pragma unroll
        for (int j = 0; j < CV; j += 4) {
            float4 v = *(const float4*)(src + j);
            aTile[r * LDA + c0 + j + 0] = f2bf(v.x);
            aTile[r * LDA + c0 + j + 1] = f2bf(v.y);
            aTile[r * LDA + c0 + j + 2] = f2bf(v.z);
            aTile[r * LDA + c0 + j + 3] = f2bf(v.w);
        }
    }
#endif
    __syncthreads();

    // ---- per-wave 16x16 output tile, K loop of 16x16x32 WMMAs ----
    const int wave = threadIdx.x >> 5;          // = N tile index
    const int lane = threadIdx.x & 31;
    const int m    = lane & 15;
    const int kh   = lane >> 4;
    const int ncol = wave * 16 + (lane & 15);

    v8f acc = {};
    for (int k0 = 0; k0 < K; k0 += 32) {
        // A fragment: two contiguous 16B groups -> 2x ds_load_b128
        const v8bf alo = *(const v8bf*)&aTile[m * LDA + k0 + kh * 16];
        const v8bf ahi = *(const v8bf*)&aTile[m * LDA + k0 + kh * 16 + 8];
        const v16bf a = __builtin_shufflevector(alo, ahi,
            0, 1, 2, 3, 4, 5, 6, 7, 8, 9, 10, 11, 12, 13, 14, 15);
        // B fragment: contiguous 32B per lane from pre-packed weights
        const v16bf b = *(const v16bf*)(Wpk +
            ((size_t)((k0 >> 5) * (N / 16) + wave) * 512) + lane * 16);
        acc = __builtin_amdgcn_wmma_f32_16x16x32_bf16(
                  /*neg_a=*/false, a, /*neg_b=*/false, b,
                  /*c_mod=*/(short)0, acc, /*reuse_a=*/false, /*reuse_b=*/false);
    }

    // ---- epilogue: bias add + store (D layout: VGPR r -> row r + kh*8) ----
    const float bv = bias[ncol];
    #pragma unroll
    for (int r = 0; r < 8; ++r)
        out[(size_t)(m0 + kh * 8 + r) * N + ncol] = acc[r] + bv;
}

// ---------------------------------------------------------------------------
// Sparse COO aggregation: out[row[e], :] += support[col[e], :] * ew[e]
// One wave per edge; lane covers DIM/32 contiguous floats. Edge index is
// wave-uniform (forced scalar via readfirstlane), so row/col/ew become
// scalar loads and addresses are SGPR-based.
// ---------------------------------------------------------------------------
template <int DIM>
__global__ __launch_bounds__(256)
void gcn_aggregate(const float* __restrict__ support,
                   const float* __restrict__ ew,
                   const int*   __restrict__ row,
                   const int*   __restrict__ col,
                   float* __restrict__ out,
                   int nEdges) {
    constexpr int VEC = DIM / 32;                // floats per lane (4 or 2)
    const int wave = threadIdx.x >> 5;
    const int lane = threadIdx.x & 31;
    int edge = __builtin_amdgcn_readfirstlane(blockIdx.x * 8 + wave);
    if (edge >= nEdges) return;

    if (lane == 0) {                             // stream edge arrays ahead
        __builtin_prefetch(col + edge + 4096, 0, 1);
        __builtin_prefetch(row + edge + 4096, 0, 1);
        __builtin_prefetch(ew  + edge + 4096, 0, 1);
    }

    const int   c = col[edge];                   // wave-uniform (scalar) loads
    const int   r = row[edge];
    const float w = ew[edge];

    const float* src = support + (size_t)c * DIM + lane * VEC;
    float*       dst = out     + (size_t)r * DIM + lane * VEC;
    float v[VEC];
    #pragma unroll
    for (int j = 0; j < VEC; ++j) v[j] = src[j]; // merges to b128/b64 load
    #pragma unroll
    for (int j = 0; j < VEC; ++j) atomicAdd(dst + j, v[j] * w);
}

// ---------------------------------------------------------------------------
// ReLU + dropout (p=0.5, scale 2.0), deterministic per-element hash mask.
// ---------------------------------------------------------------------------
__global__ void gcn_relu_dropout(float* __restrict__ h, int n) {
    const int i = blockIdx.x * blockDim.x + threadIdx.x;
    if (i >= n) return;
    float v = fmaxf(h[i], 0.0f);
    h[i] = (mix_hash((uint32_t)i) & 1u) ? v * 2.0f : 0.0f;
}

__global__ void gcn_zero_f32(float* __restrict__ p, int n) {
    const int i = blockIdx.x * blockDim.x + threadIdx.x;
    if (i < n) p[i] = 0.0f;
}

// ---------------------------------------------------------------------------
// Host-side launch
// ---------------------------------------------------------------------------
extern "C" void kernel_launch(void* const* d_in, const int* in_sizes, int n_in,
                              void* d_out, int out_size, void* d_ws, size_t ws_size,
                              hipStream_t stream) {
    (void)in_sizes; (void)n_in; (void)out_size; (void)ws_size;

    const float* x  = (const float*)d_in[0];   // [N_NODES, NFEAT]
    const float* ew = (const float*)d_in[1];   // [N_EDGES]
    const float* W1 = (const float*)d_in[2];   // [NFEAT, NHID]
    const float* b1 = (const float*)d_in[3];   // [NHID]
    const float* W2 = (const float*)d_in[4];   // [NHID, NCLASS]
    const float* b2 = (const float*)d_in[5];   // [NCLASS]
    const int*  row = (const int*)d_in[6];     // [N_EDGES]
    const int*  col = (const int*)d_in[7];     // [N_EDGES]
    float* out = (float*)d_out;                // [N_NODES, NCLASS]

    // workspace: W1pk | W2pk | support1 | h | support2   (~128.2 MB)
    __bf16* W1pk    = (__bf16*)d_ws;                          // NFEAT*NHID bf16
    __bf16* W2pk    = W1pk + (size_t)NFEAT * NHID;            // NHID*NCLASS bf16
    float*  support1 = (float*)(W2pk + (size_t)NHID * NCLASS);
    float*  h        = support1 + (size_t)N_NODES * NHID;
    float*  support2 = h        + (size_t)N_NODES * NHID;

    const int mtiles     = N_NODES / 16;        // 6250, exact
    const int edgeBlocks = (N_EDGES + 7) / 8;   // 8 waves (edges) per block

    // 0) pre-pack weights into WMMA B-fragment bf16 layout
    gcn_pack_w<NFEAT, NHID><<<(NFEAT * NHID + 255) / 256, 256, 0, stream>>>(W1, W1pk);
    gcn_pack_w<NHID, NCLASS><<<(NHID * NCLASS + 255) / 256, 256, 0, stream>>>(W2, W2pk);

    // 1) support1 = x @ W1 + b1      (WMMA bf16 -> f32)
    gcn_gemm_wmma<NFEAT, NHID, 8><<<mtiles, 256, 0, stream>>>(x, W1pk, b1, support1);

    // 2) h = A @ support1  (zero, then edge scatter-add)
    {
        const int n = N_NODES * NHID;
        gcn_zero_f32<<<(n + 255) / 256, 256, 0, stream>>>(h, n);
        gcn_aggregate<NHID><<<edgeBlocks, 256, 0, stream>>>(
            support1, ew, row, col, h, N_EDGES);
    }

    // 3) h = dropout(relu(h))
    {
        const int n = N_NODES * NHID;
        gcn_relu_dropout<<<(n + 255) / 256, 256, 0, stream>>>(h, n);
    }

    // 4) support2 = h @ W2 + b2      (WMMA bf16 -> f32)
    gcn_gemm_wmma<NHID, NCLASS, 4><<<mtiles, 128, 0, stream>>>(h, W2pk, b2, support2);

    // 5) out = A @ support2
    {
        const int n = N_NODES * NCLASS;
        gcn_zero_f32<<<(n + 255) / 256, 256, 0, stream>>>(out, n);
        gcn_aggregate<NCLASS><<<edgeBlocks, 256, 0, stream>>>(
            support2, ew, row, col, out, N_EDGES);
    }
}
